// PhaseBiasedCrossAttention_15418932592804
// MI455X (gfx1250) — matrix operand order repaired and checked by reading, not compiled, not run on hardware
//
#include <hip/hip_runtime.h>
#include <hip/hip_bf16.h>

// ---------------------------------------------------------------------------
// Problem constants
// ---------------------------------------------------------------------------
#define D_MODEL 512
#define N_HEADS 8
#define D_K     64
#define BATCH   4
#define LQ      2048
#define LK      2048
#define BL      (BATCH * LQ)          // 8192 rows for projections
#define BH      (BATCH * N_HEADS)     // 32
#define SROW    2052                  // padded LDS row stride (floats): 8*SROW%64==32

typedef __attribute__((ext_vector_type(16))) __bf16 v16bf;
typedef __attribute__((ext_vector_type(8)))  __bf16 v8bf;
typedef __attribute__((ext_vector_type(4)))  __bf16 v4bf;
typedef __attribute__((ext_vector_type(8)))  float  v8f;
typedef __attribute__((ext_vector_type(4)))  float  v4f;

static __device__ __forceinline__ v16bf cat8(v8bf lo, v8bf hi) {
    return __builtin_shufflevector(lo, hi, 0,1,2,3,4,5,6,7,8,9,10,11,12,13,14,15);
}
static __device__ __forceinline__ v8f cat4f(v4f lo, v4f hi) {
    return __builtin_shufflevector(lo, hi, 0,1,2,3,4,5,6,7);
}

static __device__ __forceinline__ v8f wmma_bf16(v16bf a, v16bf b, v8f c) {
    return __builtin_amdgcn_wmma_f32_16x16x32_bf16(
        /*neg_a=*/false, a, /*neg_b=*/false, b,
        /*c_mod=*/(short)0, c, /*reuse_a=*/false, /*reuse_b=*/false);
}

// Async copy of one 16-byte chunk global -> LDS (tracked by ASYNCcnt).
// VDST = per-lane LDS byte address, VADDR = per-lane 64-bit global address.
static __device__ __forceinline__ void async_g2l_b128(unsigned lds_byte_addr,
                                                      const void* gaddr) {
    asm volatile("global_load_async_to_lds_b128 %0, %1, off"
                 :: "v"(lds_byte_addr), "v"(gaddr) : "memory");
}
static __device__ __forceinline__ void wait_async0() {
    asm volatile("s_wait_asynccnt 0x0" ::: "memory");
}

// ---------------------------------------------------------------------------
// fp32 -> bf16 elementwise convert (4 elems / thread).
// Source is read exactly once in the whole pipeline -> non-temporal load
// (keeps the 192 MB L2 free for the reused bf16 operands).
// ---------------------------------------------------------------------------
__global__ void cvt_f32_bf16_kernel(const float* __restrict__ src,
                                    __bf16* __restrict__ dst, int n4) {
    int i = blockIdx.x * blockDim.x + threadIdx.x;
    if (i >= n4) return;
    const v4f f = __builtin_nontemporal_load((const v4f*)(src + (size_t)i * 4));
    *(v4bf*)(dst + (size_t)i * 4) = __builtin_convertvector(f, v4bf);
}

// ---------------------------------------------------------------------------
// Projection GEMM:  C[M=8192, N=512] = A[8192,512] @ W^T + bias
//   The block's 64x512 W tile (64 KB bf16) is staged once into LDS via
//   global_load_async_to_lds_b128; B fragments then come from LDS.
//   MODE 0: store bf16 head-split  [B,H,L,64]      (Q, K projections)
//   MODE 1: store bf16 transposed  [B,H,64,L]      (V projection)
//   MODE 2: store fp32 flat        [M,512]         (final out, NT store)
// Block = 256 thr (8 waves); wave computes 16(M) x 64(N); block = 128M x 64N.
// ---------------------------------------------------------------------------
template <int MODE>
__global__ __launch_bounds__(256) void proj_gemm_kernel(
        const __bf16* __restrict__ A, const __bf16* __restrict__ W,
        const float* __restrict__ bias,
        __bf16* __restrict__ outb, float* __restrict__ outf) {
    __shared__ __bf16 Wl[64 * D_MODEL];          // 64 KB

    const int tid  = threadIdx.x;
    const int wave = tid >> 5;
    const int lane = tid & 31;
    const int l16  = lane & 15;
    const int hb   = lane >> 4;

    const int n0 = blockIdx.x * 64;
    const int m0 = blockIdx.y * 128 + wave * 16;

    // ---- async-stage W rows [n0, n0+64) into LDS: 64 KB, 256 B per thread --
    {
        const char* gsrc = (const char*)(W + (size_t)n0 * D_MODEL);
        const unsigned lbase = (unsigned)(uintptr_t)(&Wl[0]);
        #pragma unroll
        for (int i = 0; i < 16; ++i) {
            const unsigned off = (unsigned)(tid * 16 + i * 4096);
            async_g2l_b128(lbase + off, gsrc + off);
        }
        wait_async0();
    }
    __syncthreads();

    v8f acc[4] = {};
    for (int k0 = 0; k0 < D_MODEL; k0 += 64) {
        // two A fragments (k0, k0+32) fetched up-front to overlap latency
        const __bf16* arow = A + (size_t)(m0 + l16) * D_MODEL + k0;
        v16bf af0 = *(const v16bf*)(arow + hb * 16);
        v16bf af1 = *(const v16bf*)(arow + 32 + hb * 16);
        #pragma unroll
        for (int t = 0; t < 4; ++t) {
            const __bf16* brow = Wl + (size_t)(t * 16 + l16) * D_MODEL + k0 + hb * 16;
            acc[t] = wmma_bf16(af0, *(const v16bf*)brow, acc[t]);
        }
        #pragma unroll
        for (int t = 0; t < 4; ++t) {
            const __bf16* brow = Wl + (size_t)(t * 16 + l16) * D_MODEL + k0 + 32 + hb * 16;
            acc[t] = wmma_bf16(af1, *(const v16bf*)brow, acc[t]);
        }
    }

    #pragma unroll
    for (int t = 0; t < 4; ++t) {
        const int n  = n0 + t * 16 + l16;
        const float bv = bias[n];
        #pragma unroll
        for (int r = 0; r < 8; ++r) {
            const int m   = m0 + r + 8 * hb;
            const float v = acc[t][r] + bv;
            if (MODE == 2) {
                __builtin_nontemporal_store(v, outf + (size_t)m * D_MODEL + n);
            } else {
                const int b = m >> 11, l = m & (LQ - 1);
                const int h = n >> 6,  d = n & (D_K - 1);
                if (MODE == 0)
                    outb[((size_t)(b * N_HEADS + h) * LQ + l) * D_K + d] = (__bf16)v;
                else
                    outb[((size_t)(b * N_HEADS + h) * D_K + d) * LK + l] = (__bf16)v;
            }
        }
    }
}

// ---------------------------------------------------------------------------
// Fused attention:  per block = one (bh, 16-q-row tile):
//   1) logits 16x2048 (QK^T/8 + cos bias) -> LDS
//   2) row softmax in LDS (exp kept unnormalized, 1/sum kept per row)
//   3) single normalized NT write of probs to global attn (required output;
//      never re-read -> TH=NT keeps L2 for Qp/Kp/Vt)
//   4) P@V from LDS probs (packed bf16 conversion) -> per-wave partials
//   5) cross-wave LDS reduction, scale by 1/sum, store bf16 ctx
// Block 256 thr (8 waves); wave owns a 256-wide k slice.
// Dynamic LDS: S[16*SROW] + R[8*16*64] + red[256] + rstat[16]  (~162 KB)
// ---------------------------------------------------------------------------
__global__ __launch_bounds__(256) void attn_fused_kernel(
        const __bf16* __restrict__ Qp, const __bf16* __restrict__ Kp,
        const __bf16* __restrict__ Vt, const float* __restrict__ log_omega,
        float* __restrict__ attn, __bf16* __restrict__ ctx) {
    extern __shared__ float smem[];
    float* S     = smem;                  // 16 * SROW
    float* R     = S + 16 * SROW;         // 8 * 16 * 64
    float* red   = R + 8 * 16 * 64;       // 256
    float* rstat = red + 256;             // 16  (row max, then 1/sum)

    const int tid  = threadIdx.x;
    const int wave = tid >> 5;
    const int lane = tid & 31;
    const int l16  = lane & 15;
    const int hb   = lane >> 4;

    const int q0 = blockIdx.x * 16;
    const int bh = blockIdx.y;
    const int h  = bh & (N_HEADS - 1);
    const float twopiw = 6.28318530717958647692f * __expf(log_omega[h]);

    // ---------------- Phase 1: logits tile into LDS -------------------------
    const __bf16* Qbase = Qp + ((size_t)bh * LQ + q0) * D_K;
    const __bf16* Kbase = Kp + (size_t)bh * LK * D_K;

    v16bf afrag[2];
    #pragma unroll
    for (int s = 0; s < 2; ++s) {
        const __bf16* arow = Qbase + (size_t)l16 * D_K + s * 32;
        afrag[s] = *(const v16bf*)(arow + hb * 16);
    }
    const int kw0 = wave * 256;
    #pragma unroll
    for (int t = 0; t < 16; ++t) {
        const int kcol0 = kw0 + t * 16;
        v8f acc = {};
        #pragma unroll
        for (int s = 0; s < 2; ++s) {
            const __bf16* brow = Kbase + (size_t)(kcol0 + l16) * D_K + s * 32 + hb * 16;
            acc = wmma_bf16(afrag[s], *(const v16bf*)brow, acc);
        }
        const int kcol = kcol0 + l16;
        #pragma unroll
        for (int r = 0; r < 8; ++r) {
            const int qr = r + 8 * hb;
            const float pb = __cosf(twopiw * (float)(q0 + qr - kcol));
            S[qr * SROW + kcol] = acc[r] * 0.125f + pb;
        }
    }
    __syncthreads();

    // ---------------- Phase 2: softmax statistics in LDS --------------------
    {
        const int row = tid >> 4, seg = tid & 15;
        float* p = S + row * SROW + seg * 128;
        float m = -3.4e38f;
        for (int i = 0; i < 128; i += 4) {
            const v4f v = *(const v4f*)(p + i);
            m = fmaxf(m, fmaxf(fmaxf(v.x, v.y), fmaxf(v.z, v.w)));
        }
        red[tid] = m;
        __syncthreads();
        if (seg == 0) {
            float mm = red[tid];
            #pragma unroll
            for (int j = 1; j < 16; ++j) mm = fmaxf(mm, red[tid + j]);
            rstat[row] = mm;
        }
        __syncthreads();
        const float rm = rstat[row];
        float sum = 0.f;
        for (int i = 0; i < 128; i += 4) {
            v4f v = *(const v4f*)(p + i);
            v.x = __expf(v.x - rm); v.y = __expf(v.y - rm);
            v.z = __expf(v.z - rm); v.w = __expf(v.w - rm);
            *(v4f*)(p + i) = v;
            sum += v.x + v.y + v.z + v.w;
        }
        red[tid] = sum;
        __syncthreads();
        if (seg == 0) {
            float ss = red[tid];
            #pragma unroll
            for (int j = 1; j < 16; ++j) ss += red[tid + j];
            rstat[row] = __frcp_rn(ss);       // overwrite max with 1/sum
        }
        __syncthreads();
    }

    // ---------------- Phase 3: single normalized NT write of probs ----------
    {
        float* arow = attn + ((size_t)bh * LQ + q0) * LK;
        for (int r = 0; r < 16; ++r) {
            const float inv = rstat[r];
            const float* sp = S + r * SROW + tid * 8;
            const v4f a = *(const v4f*)(sp)     * inv;
            const v4f b = *(const v4f*)(sp + 4) * inv;
            float* dp = arow + (size_t)r * LK + tid * 8;
            __builtin_nontemporal_store(a, (v4f*)(dp));
            __builtin_nontemporal_store(b, (v4f*)(dp + 4));
        }
    }

    // ---------------- Phase 4: P@V partials (wave-local 256-k slice) --------
    const __bf16* Vb = Vt + (size_t)bh * D_K * LK;
    v8f acc[4] = {};
    for (int kc = 0; kc < 256; kc += 32) {
        const float* prow = S + l16 * SROW + kw0 + kc;
        const v8f f0 = cat4f(*(const v4f*)(prow + hb * 16),
                             *(const v4f*)(prow + hb * 16 + 4));
        const v8f f1 = cat4f(*(const v4f*)(prow + hb * 16 + 8),
                             *(const v4f*)(prow + hb * 16 + 12));
        const v16bf af = cat8(__builtin_convertvector(f0, v8bf),
                              __builtin_convertvector(f1, v8bf));
        #pragma unroll
        for (int t = 0; t < 4; ++t) {
            const __bf16* brow = Vb + (size_t)(t * 16 + l16) * LK + kw0 + kc + hb * 16;
            acc[t] = wmma_bf16(af, *(const v16bf*)brow, acc[t]);
        }
    }
    #pragma unroll
    for (int t = 0; t < 4; ++t) {
        const int d = t * 16 + l16;
        #pragma unroll
        for (int r = 0; r < 8; ++r) {
            const int qr = r + 8 * hb;
            R[wave * 1024 + qr * 64 + d] = acc[t][r];
        }
    }
    __syncthreads();

    // ---------------- Phase 5: cross-wave reduce + scaled bf16 ctx store ----
    {
        const int e0 = tid * 4;                 // 1024 elems / 256 thr
        const int qr = e0 >> 6, d = e0 & 63;
        v4f s = {0.f, 0.f, 0.f, 0.f};
        #pragma unroll
        for (int w = 0; w < 8; ++w) s += *(const v4f*)(R + w * 1024 + e0);
        s *= rstat[qr];
        const int b = bh >> 3;
        *(v4bf*)(ctx + ((size_t)b * LQ + q0 + qr) * D_MODEL + h * D_K + d) =
            __builtin_convertvector(s, v4bf);
    }
}

// ---------------------------------------------------------------------------
// Launcher
// ---------------------------------------------------------------------------
extern "C" void kernel_launch(void* const* d_in, const int* in_sizes, int n_in,
                              void* d_out, int out_size, void* d_ws, size_t ws_size,
                              hipStream_t stream) {
    (void)in_sizes; (void)n_in; (void)out_size; (void)ws_size;

    const float* Q_in = (const float*)d_in[0];
    const float* K_in = (const float*)d_in[1];
    const float* V_in = (const float*)d_in[2];
    const float* Wq   = (const float*)d_in[3];
    const float* bq   = (const float*)d_in[4];
    const float* Wk   = (const float*)d_in[5];
    const float* bk   = (const float*)d_in[6];
    const float* Wv   = (const float*)d_in[7];
    const float* bv   = (const float*)d_in[8];
    const float* Wo   = (const float*)d_in[9];
    const float* bo   = (const float*)d_in[10];
    const float* lw   = (const float*)d_in[11];

    float* out  = (float*)d_out;
    float* attn = out + (size_t)BATCH * LQ * D_MODEL;

    // Workspace layout (bf16)
    __bf16* Xq  = (__bf16*)d_ws;
    __bf16* Xk  = Xq  + (size_t)BL * D_MODEL;
    __bf16* Xv  = Xk  + (size_t)BL * D_MODEL;
    __bf16* Wqb = Xv  + (size_t)BL * D_MODEL;
    __bf16* Wkb = Wqb + (size_t)D_MODEL * D_MODEL;
    __bf16* Wvb = Wkb + (size_t)D_MODEL * D_MODEL;
    __bf16* Wob = Wvb + (size_t)D_MODEL * D_MODEL;
    __bf16* Qp  = Wob + (size_t)D_MODEL * D_MODEL;   // [BH, L, 64]
    __bf16* Kp  = Qp  + (size_t)BH * LQ * D_K;       // [BH, L, 64]
    __bf16* Vt  = Kp  + (size_t)BH * LK * D_K;       // [BH, 64, L]
    __bf16* ctx = Vt  + (size_t)BH * D_K * LK;       // [B, L, 512]

    const int XN4 = (BL * D_MODEL) / 4;
    const int WN4 = (D_MODEL * D_MODEL) / 4;
    cvt_f32_bf16_kernel<<<(XN4 + 255) / 256, 256, 0, stream>>>(Q_in, Xq, XN4);
    cvt_f32_bf16_kernel<<<(XN4 + 255) / 256, 256, 0, stream>>>(K_in, Xk, XN4);
    cvt_f32_bf16_kernel<<<(XN4 + 255) / 256, 256, 0, stream>>>(V_in, Xv, XN4);
    cvt_f32_bf16_kernel<<<(WN4 + 255) / 256, 256, 0, stream>>>(Wq, Wqb, WN4);
    cvt_f32_bf16_kernel<<<(WN4 + 255) / 256, 256, 0, stream>>>(Wk, Wkb, WN4);
    cvt_f32_bf16_kernel<<<(WN4 + 255) / 256, 256, 0, stream>>>(Wv, Wvb, WN4);
    cvt_f32_bf16_kernel<<<(WN4 + 255) / 256, 256, 0, stream>>>(Wo, Wob, WN4);

    const dim3 gproj(D_MODEL / 64, BL / 128);        // (8, 64)
    proj_gemm_kernel<0><<<gproj, 256, 0, stream>>>(Xq, Wqb, bq, Qp, nullptr);
    proj_gemm_kernel<0><<<gproj, 256, 0, stream>>>(Xk, Wkb, bk, Kp, nullptr);
    proj_gemm_kernel<1><<<gproj, 256, 0, stream>>>(Xv, Wvb, bv, Vt, nullptr);

    const size_t smem_bytes = (size_t)(16 * SROW + 8 * 16 * 64 + 256 + 16) * sizeof(float);
    attn_fused_kernel<<<dim3(LQ / 16, BH), 256, smem_bytes, stream>>>(
        Qp, Kp, Vt, lw, attn, ctx);

    proj_gemm_kernel<2><<<gproj, 256, 0, stream>>>(ctx, Wob, bo, nullptr, out);
}
